// HybridVisionTransformerAttention_55259049230701
// MI455X (gfx1250) — compile-verified, matrix operand order checked by reading, not run on hardware
//
#include <hip/hip_runtime.h>

// ---------------------------------------------------------------------------
// Hybrid ViT attention block for gfx1250 (MI455X), bf16 WMMA everywhere.
// GEMMs use double-buffered LDS staging via GLOBAL_LOAD_ASYNC_TO_LDS_B128
// (ASYNCcnt-tracked, no VGPR round-trip).
// ---------------------------------------------------------------------------

#define CC     384
#define C4     1536
#define NPIX   784
#define BATCH  16
#define HEADS  8
#define HD     48
#define MT     (BATCH * NPIX)   // 12544 tokens
#define NPAD   800              // 784 padded to multiple of 32
#define BN_EPS 1e-5f

typedef __attribute__((ext_vector_type(16))) __bf16 v16bf;
typedef __attribute__((ext_vector_type(8)))  __bf16 v8bf;
typedef __attribute__((ext_vector_type(8)))  float  v8f;

__device__ __forceinline__ unsigned short f2bfu(float f) {
    union { float f; unsigned u; } a; a.f = f;
    unsigned u = a.u;
    u += 0x7fffu + ((u >> 16) & 1u);   // round-to-nearest-even
    return (unsigned short)(u >> 16);
}

__device__ __forceinline__ __bf16 bfzero() {
    unsigned short z = 0; __bf16 r;
    __builtin_memcpy(&r, &z, sizeof(r));
    return r;
}

__device__ __forceinline__ v16bf cat16(v8bf lo, v8bf hi) {
    v16bf r;
#pragma unroll
    for (int i = 0; i < 8; ++i) { r[i] = lo[i]; r[i + 8] = hi[i]; }
    return r;
}

// --- CDNA5 async global->LDS (16B per lane), ASYNCcnt tracked --------------
__device__ __forceinline__ void async_load16(unsigned lds_off,
                                             const void* gaddr) {
    asm volatile("global_load_async_to_lds_b128 %0, %1, off"
                 :: "v"(lds_off),
                    "v"((unsigned long long)(uintptr_t)gaddr)
                 : "memory");
}
__device__ __forceinline__ void wait_async0() {
#if __has_builtin(__builtin_amdgcn_s_wait_asynccnt)
    __builtin_amdgcn_s_wait_asynccnt(0);
#else
    asm volatile("s_wait_asynccnt 0x0" ::: "memory");
#endif
}
__device__ __forceinline__ void wait_async6() {
#if __has_builtin(__builtin_amdgcn_s_wait_asynccnt)
    __builtin_amdgcn_s_wait_asynccnt(6);
#else
    asm volatile("s_wait_asynccnt 0x6" ::: "memory");
#endif
}

// ---------------------------------------------------------------------------
// Elementwise helpers
// ---------------------------------------------------------------------------
__global__ __launch_bounds__(256) void f32_to_bf16_kernel(
    const float* __restrict__ in, unsigned short* __restrict__ out, int n) {
    int i = blockIdx.x * 256 + threadIdx.x;
    if (i < n) out[i] = f2bfu(in[i]);
}

__global__ __launch_bounds__(256) void zero_u16_kernel(
    unsigned short* __restrict__ p, int n) {
    int i = blockIdx.x * 256 + threadIdx.x;
    if (i < n) p[i] = 0;
}

// Depthwise 3x3 conv (SAME) + bias + BN1; NCHW input -> [token][chan] f32+bf16
__global__ __launch_bounds__(256) void conv_bn_kernel(
    const float* __restrict__ x, const float* __restrict__ w,
    const float* __restrict__ cb,
    const float* __restrict__ g, const float* __restrict__ bb,
    const float* __restrict__ m, const float* __restrict__ v,
    float* __restrict__ lxf, unsigned short* __restrict__ lxb) {
    int idx = blockIdx.x * 256 + threadIdx.x;            // over MT*CC, [t][c]
    if (idx >= MT * CC) return;
    int c = idx % CC;
    int t = idx / CC;
    int b = t / NPIX, n = t % NPIX;
    int hy = n / 28, wx = n % 28;
    const float* xb = x + ((size_t)b * CC + c) * NPIX;
    const float* wc = w + c * 9;
    float s = 0.f;
#pragma unroll
    for (int dy = -1; dy <= 1; ++dy) {
#pragma unroll
        for (int dx = -1; dx <= 1; ++dx) {
            int yy = hy + dy, xx = wx + dx;
            if (yy >= 0 && yy < 28 && xx >= 0 && xx < 28)
                s += xb[yy * 28 + xx] * wc[(dy + 1) * 3 + (dx + 1)];
        }
    }
    s += cb[c];
    s = (s - m[c]) * rsqrtf(v[c] + BN_EPS) * g[c] + bb[c];
    lxf[idx] = s;
    lxb[idx] = f2bfu(s);
}

// ---------------------------------------------------------------------------
// WMMA bf16 GEMM:  C[M,Nn] = A[M,K] * W[Nn,K]^T  (+ epilogue)
//   Block: 128 threads (4 waves). Block tile 128x64, wave tile 32x64
//   (8 WMMAs / wave / K-step). K-step 32 via double-buffered async LDS.
// EPI: 0 = bias -> bf16 out [M,Nn]
//      1 = bias -> bf16 scattered as V^T [b,h,d,NPAD]
//      2 = bias + residual -> f32 out; BN2 -> bf16 out          (proj)
//      3 = bias + exact GELU -> bf16 out                        (ffn1)
//      4 = bias + residual -> f32 out in NCHW                   (ffn2)
// ---------------------------------------------------------------------------
template <int EPI>
__global__ __launch_bounds__(128) void gemm_bf16(
    const unsigned short* __restrict__ A,   // [M,K] bf16
    const unsigned short* __restrict__ W,   // [Nn,K] bf16 (orig weight layout)
    const float* __restrict__ bias,         // [Nn]
    const float* __restrict__ res,          // [M,Nn] f32 (EPI 2,4)
    float* __restrict__ outf,               // f32 out (EPI 2,4)
    unsigned short* __restrict__ outb,      // bf16 out (EPI 0,1,2,3)
    const float* __restrict__ bn_g, const float* __restrict__ bn_b,
    const float* __restrict__ bn_m, const float* __restrict__ bn_v,
    int M, int Nn, int K) {
    __shared__ __align__(16) unsigned short As[2][128 * 32];
    __shared__ __align__(16) unsigned short Bs[2][64 * 32];

    const int tid  = threadIdx.x;
    const int wave = tid >> 5;
    const int lane = tid & 31;
    const int mloc = lane & 15;
    const int kk   = (lane >> 4) * 8;       // K sub-offset / row half
    const int rowBlk = blockIdx.x * 128;
    const int colBlk = blockIdx.y * 64;

    v8f zero = {};
    v8f acc[2][4];
#pragma unroll
    for (int s = 0; s < 2; ++s)
#pragma unroll
        for (int t = 0; t < 4; ++t) acc[s][t] = zero;

    // issue 6 async 16B loads per thread for one K-slab into buffer `buf`
    auto stage = [&](int buf, int k0) {
        // A: 128 rows x 32 halves = 512 chunks of 8 halves; 4 per thread
#pragma unroll
        for (int j = 0; j < 4; ++j) {
            int c = tid * 4 + j;
            int r = c >> 2, c8 = (c & 3) * 8;
            async_load16((unsigned)(uintptr_t)&As[buf][r * 32 + c8],
                         &A[(size_t)(rowBlk + r) * K + k0 + c8]);
        }
        // B: 64 cols x 32 halves = 256 chunks; 2 per thread
#pragma unroll
        for (int j = 0; j < 2; ++j) {
            int c = tid * 2 + j;
            int r = c >> 2, c8 = (c & 3) * 8;
            async_load16((unsigned)(uintptr_t)&Bs[buf][r * 32 + c8],
                         &W[(size_t)(colBlk + r) * K + k0 + c8]);
        }
    };

    const int nk = K / 32;
    stage(0, 0);
    for (int ks = 0; ks < nk; ++ks) {
        const int cur = ks & 1;
        if (ks + 1 < nk) {
            stage(cur ^ 1, (ks + 1) * 32);  // prefetch next slab
            wait_async6();                  // current slab's 6 loads done
        } else {
            wait_async0();
        }
        __syncthreads();                    // all threads' slab visible

#pragma unroll
        for (int s = 0; s < 2; ++s) {
            const int arow = (wave * 32 + s * 16 + mloc) * 32;
            v16bf afrag = cat16(*(const v8bf*)&As[cur][arow + kk],
                                *(const v8bf*)&As[cur][arow + kk + 16]);
#pragma unroll
            for (int t = 0; t < 4; ++t) {
                const int bcol = (t * 16 + mloc) * 32;
                v16bf bfrag = cat16(*(const v8bf*)&Bs[cur][bcol + kk],
                                    *(const v8bf*)&Bs[cur][bcol + kk + 16]);
                acc[s][t] = __builtin_amdgcn_wmma_f32_16x16x32_bf16(
                    false, afrag, false, bfrag, (short)0, acc[s][t],
                    false, false);
            }
        }
        __syncthreads();                    // everyone done reading slab
    }

    // epilogue
    const int rhalf = (lane >> 4) * 8;
#pragma unroll
    for (int s = 0; s < 2; ++s) {
        const int mrowBase = rowBlk + wave * 32 + s * 16 + rhalf;
#pragma unroll
        for (int t = 0; t < 4; ++t) {
            int ncol = colBlk + t * 16 + mloc;
            float bia = bias[ncol];
#pragma unroll
            for (int r = 0; r < 8; ++r) {
                int row = mrowBase + r;
                float val = acc[s][t][r] + bia;
                if (EPI == 0) {
                    outb[(size_t)row * Nn + ncol] = f2bfu(val);
                } else if (EPI == 1) {
                    int bb_ = row / NPIX, n_ = row % NPIX;
                    int h_ = ncol / HD, d_ = ncol % HD;
                    outb[((size_t)(bb_ * HEADS + h_) * HD + d_) * NPAD + n_] =
                        f2bfu(val);
                } else if (EPI == 2) {
                    float y = val + res[(size_t)row * Nn + ncol];
                    outf[(size_t)row * Nn + ncol] = y;
                    float yb = (y - bn_m[ncol]) *
                                   rsqrtf(bn_v[ncol] + BN_EPS) * bn_g[ncol] +
                               bn_b[ncol];
                    outb[(size_t)row * Nn + ncol] = f2bfu(yb);
                } else if (EPI == 3) {
                    float gl = 0.5f * val *
                               (1.0f + erff(val * 0.70710678118654752f));
                    outb[(size_t)row * Nn + ncol] = f2bfu(gl);
                } else {
                    float o = val + res[(size_t)row * Nn + ncol];
                    int bb_ = row / NPIX, n_ = row % NPIX;
                    outf[((size_t)bb_ * CC + ncol) * NPIX + n_] = o;
                }
            }
        }
    }
}

// ---------------------------------------------------------------------------
// Attention: one wave per (q-row-tile, head, batch).
// Scores via WMMA (K=48 padded to 64), 2-pass softmax, PV via WMMA with
// P staged bf16 in LDS (16 x 800).
// ---------------------------------------------------------------------------
__global__ __launch_bounds__(32) void attn_kernel(
    const unsigned short* __restrict__ qb,   // [tok][chan] bf16
    const unsigned short* __restrict__ kb,   // [tok][chan] bf16
    const unsigned short* __restrict__ vt,   // [b,h,d,NPAD] bf16 (tail zeroed)
    unsigned short* __restrict__ ob) {       // [tok][chan] bf16
    __shared__ __align__(16) unsigned short P[16 * NPAD];

    const int rt = blockIdx.x;     // 0..48
    const int h  = blockIdx.y;
    const int b  = blockIdx.z;
    const int lane  = threadIdx.x;
    const int mloc  = lane & 15;
    const int kk    = (lane >> 4) * 8;
    const int rhalf = kk;          // row half offset for C fragments
    const int rowBase = rt * 16;
    const int tokBase = b * NPIX;
    const int cbase   = h * HD;
    const float scale = 0.14433756729740643f;   // 1/sqrt(48)

    v8f zero = {};

    // Q fragments (K = 0..31 in a0, 32..63 in a1 with 48..63 zero-padded)
    v16bf a0, a1;
    {
        const unsigned short* qrow =
            qb + (size_t)(tokBase + rowBase + mloc) * CC + cbase;
        v8bf q0 = *(const v8bf*)(qrow + kk);
        v8bf q1 = *(const v8bf*)(qrow + 16 + kk);
        v8bf q2 = *(const v8bf*)(qrow + 32 + kk);
        __bf16 z = bfzero();
#pragma unroll
        for (int i = 0; i < 8; ++i) {
            a0[i] = q0[i]; a0[i + 8] = q1[i];
            a1[i] = q2[i]; a1[i + 8] = z;
        }
    }

    // ---- pass 1: row max ----
    float rm[8];
#pragma unroll
    for (int r = 0; r < 8; ++r) rm[r] = -1e30f;

    for (int ct = 0; ct < 49; ++ct) {
        const unsigned short* krow =
            kb + (size_t)(tokBase + ct * 16 + mloc) * CC + cbase;
        v8bf k0v = *(const v8bf*)(krow + kk);
        v8bf k1v = *(const v8bf*)(krow + 16 + kk);
        v8bf k2v = *(const v8bf*)(krow + 32 + kk);
        __bf16 z = bfzero();
        v16bf b0, b1;
#pragma unroll
        for (int i = 0; i < 8; ++i) {
            b0[i] = k0v[i]; b0[i + 8] = k1v[i];
            b1[i] = k2v[i]; b1[i + 8] = z;
        }
        v8f acc = zero;
        acc = __builtin_amdgcn_wmma_f32_16x16x32_bf16(
            false, a0, false, b0, (short)0, acc, false, false);
        acc = __builtin_amdgcn_wmma_f32_16x16x32_bf16(
            false, a1, false, b1, (short)0, acc, false, false);
#pragma unroll
        for (int r = 0; r < 8; ++r) rm[r] = fmaxf(rm[r], acc[r] * scale);
    }
#pragma unroll
    for (int msk = 1; msk < 16; msk <<= 1) {
#pragma unroll
        for (int r = 0; r < 8; ++r)
            rm[r] = fmaxf(rm[r], __shfl_xor(rm[r], msk, 32));
    }

    // ---- pass 2: exp -> P (bf16 in LDS), row sums ----
    float rs[8];
#pragma unroll
    for (int r = 0; r < 8; ++r) rs[r] = 0.f;

    for (int ct = 0; ct < 49; ++ct) {
        const unsigned short* krow =
            kb + (size_t)(tokBase + ct * 16 + mloc) * CC + cbase;
        v8bf k0v = *(const v8bf*)(krow + kk);
        v8bf k1v = *(const v8bf*)(krow + 16 + kk);
        v8bf k2v = *(const v8bf*)(krow + 32 + kk);
        __bf16 z = bfzero();
        v16bf b0, b1;
#pragma unroll
        for (int i = 0; i < 8; ++i) {
            b0[i] = k0v[i]; b0[i + 8] = k1v[i];
            b1[i] = k2v[i]; b1[i + 8] = z;
        }
        v8f acc = zero;
        acc = __builtin_amdgcn_wmma_f32_16x16x32_bf16(
            false, a0, false, b0, (short)0, acc, false, false);
        acc = __builtin_amdgcn_wmma_f32_16x16x32_bf16(
            false, a1, false, b1, (short)0, acc, false, false);
#pragma unroll
        for (int r = 0; r < 8; ++r) {
            float p = __expf(acc[r] * scale - rm[r]);
            rs[r] += p;
            P[(r + rhalf) * NPAD + ct * 16 + mloc] = f2bfu(p);
        }
    }
#pragma unroll
    for (int msk = 1; msk < 16; msk <<= 1) {
#pragma unroll
        for (int r = 0; r < 8; ++r) rs[r] += __shfl_xor(rs[r], msk, 32);
    }
    // zero padded columns 784..799
#pragma unroll
    for (int i = 0; i < 8; ++i) {
        int idx = lane * 8 + i;               // 256 pad entries
        P[(idx >> 4) * NPAD + NPIX + (idx & 15)] = 0;
    }
    __syncthreads();

    // ---- PV: out[16x48] = P[16x800] @ V[800x48] ----
    v8f o[3] = {zero, zero, zero};
    const size_t vbase = (size_t)(b * HEADS + h) * HD * NPAD;
    for (int ks = 0; ks < 25; ++ks) {
        const unsigned short* prow = &P[mloc * NPAD + ks * 32 + kk];
        v16bf afrag =
            cat16(*(const v8bf*)prow, *(const v8bf*)(prow + 16));
#pragma unroll
        for (int t = 0; t < 3; ++t) {
            const unsigned short* vrow =
                vt + vbase + (size_t)(t * 16 + mloc) * NPAD + ks * 32 + kk;
            v16bf bfrag =
                cat16(*(const v8bf*)vrow, *(const v8bf*)(vrow + 16));
            o[t] = __builtin_amdgcn_wmma_f32_16x16x32_bf16(
                false, afrag, false, bfrag, (short)0, o[t], false, false);
        }
    }

    // normalize by row sum, store bf16 [tok][chan]
#pragma unroll
    for (int t = 0; t < 3; ++t) {
#pragma unroll
        for (int r = 0; r < 8; ++r) {
            int row = rowBase + r + rhalf;
            int d = t * 16 + mloc;
            float val = o[t][r] / rs[r];
            ob[(size_t)(tokBase + row) * CC + cbase + d] = f2bfu(val);
        }
    }
}

// ---------------------------------------------------------------------------
// Launcher
// ---------------------------------------------------------------------------
extern "C" void kernel_launch(void* const* d_in, const int* in_sizes, int n_in,
                              void* d_out, int out_size, void* d_ws,
                              size_t ws_size, hipStream_t stream) {
    (void)in_sizes; (void)n_in; (void)out_size; (void)ws_size;
    const float* x    = (const float*)d_in[0];
    const float* lw   = (const float*)d_in[1];
    const float* lb   = (const float*)d_in[2];
    const float* bn1g = (const float*)d_in[3];
    const float* bn1b = (const float*)d_in[4];
    const float* bn1m = (const float*)d_in[5];
    const float* bn1v = (const float*)d_in[6];
    const float* qw   = (const float*)d_in[7];
    const float* qbs  = (const float*)d_in[8];
    const float* kw   = (const float*)d_in[9];
    const float* kbs  = (const float*)d_in[10];
    const float* vw   = (const float*)d_in[11];
    const float* vbs  = (const float*)d_in[12];
    const float* pw   = (const float*)d_in[13];
    const float* pbs  = (const float*)d_in[14];
    const float* f1w  = (const float*)d_in[15];
    const float* f1b  = (const float*)d_in[16];
    const float* f2w  = (const float*)d_in[17];
    const float* f2b  = (const float*)d_in[18];
    const float* bn2g = (const float*)d_in[19];
    const float* bn2b = (const float*)d_in[20];
    const float* bn2m = (const float*)d_in[21];
    const float* bn2v = (const float*)d_in[22];
    float* out = (float*)d_out;

    char* ws = (char*)d_ws;
    size_t off = 0;
    auto alloc = [&](size_t bytes) -> void* {
        void* p = ws + off;
        off = (off + bytes + 255) & ~(size_t)255;
        return p;
    };
    float*          lxf  = (float*)alloc((size_t)MT * CC * 4);
    unsigned short* lxb  = (unsigned short*)alloc((size_t)MT * CC * 2);
    unsigned short* qbuf = (unsigned short*)alloc((size_t)MT * CC * 2);
    unsigned short* kbuf = (unsigned short*)alloc((size_t)MT * CC * 2);
    unsigned short* vtb  = (unsigned short*)alloc((size_t)BATCH * HEADS * HD * NPAD * 2);
    unsigned short* aob  = (unsigned short*)alloc((size_t)MT * CC * 2);
    float*          yf   = (float*)alloc((size_t)MT * CC * 4);
    unsigned short* ybn  = (unsigned short*)alloc((size_t)MT * CC * 2);
    unsigned short* h1b  = (unsigned short*)alloc((size_t)MT * C4 * 2);
    unsigned short* qwb  = (unsigned short*)alloc((size_t)CC * CC * 2);
    unsigned short* kwb  = (unsigned short*)alloc((size_t)CC * CC * 2);
    unsigned short* vwb  = (unsigned short*)alloc((size_t)CC * CC * 2);
    unsigned short* pwb  = (unsigned short*)alloc((size_t)CC * CC * 2);
    unsigned short* f1wb = (unsigned short*)alloc((size_t)C4 * CC * 2);
    unsigned short* f2wb = (unsigned short*)alloc((size_t)CC * C4 * 2);

    // weights -> bf16
    int wn = CC * CC, w1 = C4 * CC;
    f32_to_bf16_kernel<<<(wn + 255) / 256, 256, 0, stream>>>(qw, qwb, wn);
    f32_to_bf16_kernel<<<(wn + 255) / 256, 256, 0, stream>>>(kw, kwb, wn);
    f32_to_bf16_kernel<<<(wn + 255) / 256, 256, 0, stream>>>(vw, vwb, wn);
    f32_to_bf16_kernel<<<(wn + 255) / 256, 256, 0, stream>>>(pw, pwb, wn);
    f32_to_bf16_kernel<<<(w1 + 255) / 256, 256, 0, stream>>>(f1w, f1wb, w1);
    f32_to_bf16_kernel<<<(w1 + 255) / 256, 256, 0, stream>>>(f2w, f2wb, w1);

    // depthwise conv + BN1
    conv_bn_kernel<<<(MT * CC + 255) / 256, 256, 0, stream>>>(
        x, lw, lb, bn1g, bn1b, bn1m, bn1v, lxf, lxb);

    // zero V^T (for K padding 784->800)
    int vtn = BATCH * HEADS * HD * NPAD;
    zero_u16_kernel<<<(vtn + 255) / 256, 256, 0, stream>>>(vtb, vtn);

    dim3 blk(128);
    dim3 gC(MT / 128, CC / 64);   // 98 x 6
    dim3 g4(MT / 128, C4 / 64);   // 98 x 24

    // QKV projections
    gemm_bf16<0><<<gC, blk, 0, stream>>>(lxb, qwb, qbs, nullptr, nullptr, qbuf,
                                         nullptr, nullptr, nullptr, nullptr,
                                         MT, CC, CC);
    gemm_bf16<0><<<gC, blk, 0, stream>>>(lxb, kwb, kbs, nullptr, nullptr, kbuf,
                                         nullptr, nullptr, nullptr, nullptr,
                                         MT, CC, CC);
    gemm_bf16<1><<<gC, blk, 0, stream>>>(lxb, vwb, vbs, nullptr, nullptr, vtb,
                                         nullptr, nullptr, nullptr, nullptr,
                                         MT, CC, CC);

    // multi-head attention
    attn_kernel<<<dim3(49, HEADS, BATCH), 32, 0, stream>>>(qbuf, kbuf, vtb, aob);

    // proj + residual(lx) -> yf ; BN2 -> ybn (bf16)
    gemm_bf16<2><<<gC, blk, 0, stream>>>(aob, pwb, pbs, lxf, yf, ybn,
                                         bn2g, bn2b, bn2m, bn2v, MT, CC, CC);
    // FFN1 + GELU
    gemm_bf16<3><<<g4, blk, 0, stream>>>(ybn, f1wb, f1b, nullptr, nullptr, h1b,
                                         nullptr, nullptr, nullptr, nullptr,
                                         MT, C4, CC);
    // FFN2 + residual(yf) -> NCHW f32 output
    gemm_bf16<4><<<gC, blk, 0, stream>>>(h1b, f2wb, f2b, yf, out, nullptr,
                                         nullptr, nullptr, nullptr, nullptr,
                                         MT, CC, C4);
}
